// LocalAttention_34144990003541
// MI455X (gfx1250) — matrix operand order repaired and checked by reading, not compiled, not run on hardware
//
#include <hip/hip_runtime.h>

typedef unsigned short u16;
typedef __bf16 v16bf __attribute__((ext_vector_type(16)));
typedef float  v8f   __attribute__((ext_vector_type(8)));
typedef unsigned u32x4 __attribute__((ext_vector_type(4)));
typedef unsigned u32x8 __attribute__((ext_vector_type(8)));

// ---------- helpers ----------
__device__ inline u16 f2bf(float f) {           // round-to-nearest-even
    unsigned u = __float_as_uint(f);
    unsigned r = (u + 0x7FFFu + ((u >> 16) & 1u)) >> 16;
    return (u16)r;
}

struct Frag { union { v16bf v; uint4 u[2]; }; };

// A-operand (M x 32, 16-bit): element e -> k = koff + (e<8?0:16) + hf*8 + e%8
__device__ inline Frag load_a(const u16* p, int ld, int row, int koff, int hf) {
    Frag f;
    f.u[0] = *(const uint4*)(p + row * ld + koff + hf * 8);
    f.u[1] = *(const uint4*)(p + row * ld + koff + 16 + hf * 8);
    return f;
}
// B-operand (32 x N, 16-bit), stored transposed [n][k]: element e -> k = koff + (e<8?0:16) + hf*8 + e%8
__device__ inline Frag load_b(const u16* p, int ld, int row, int koff, int hf) {
    Frag f;
    f.u[0] = *(const uint4*)(p + row * ld + koff + hf * 8);
    f.u[1] = *(const uint4*)(p + row * ld + koff + 16 + hf * 8);
    return f;
}
__device__ inline v8f wmma_bf16(const Frag& a, const Frag& b, v8f c) {
    return __builtin_amdgcn_wmma_f32_16x16x32_bf16(false, a.v, false, b.v,
                                                   (short)0, c, false, false);
}
__device__ inline void lds_wave_fence() {
    asm volatile("s_wait_dscnt 0" ::: "memory");
}
__device__ inline void async_wait0() {
    asm volatile("s_wait_asynccnt 0" ::: "memory");
}
// async copy 16B global -> LDS (per-lane), tracked by ASYNCcnt
__device__ inline void async_copy_b128(unsigned lds_byte_off, const void* gptr) {
    asm volatile("global_load_async_to_lds_b128 %0, %1, off"
                 :: "v"(lds_byte_off), "v"((unsigned long long)gptr) : "memory");
}
__device__ inline unsigned lds_off(const void* p) {   // LDS aperture: offset = addr[31:0]
    return (unsigned)(unsigned long long)p;
}

// ---------- converts ----------
__global__ void k_cvt(const float* __restrict__ in, u16* __restrict__ out, int n) {
    int i = blockIdx.x * 256 + threadIdx.x;
    if (i < n) out[i] = f2bf(in[i]);
}
// OIHW [M][C][3][3] -> [M][tap(9)][C]
__global__ void k_cvt_w(const float* __restrict__ in, u16* __restrict__ out, int M, int C) {
    int i = blockIdx.x * 256 + threadIdx.x;
    int total = M * C * 9;
    if (i >= total) return;
    int m = i / (C * 9), rem = i % (C * 9);
    int tap = rem / C, c = rem % C;
    out[i] = f2bf(in[(m * C + c) * 9 + tap]);
}

// ---------- implicit-GEMM conv3x3, pad=1, NCHW 96x96, batch 2 ----------
// MODE 0: +bias, ReLU, bf16 out.   MODE 1: +bias, +residual, f32 out.
template<int CINC, int MC, int MODE>
__global__ __launch_bounds__(256)
void conv_gemm(const u16* __restrict__ inp, const u16* __restrict__ wt,
               const float* __restrict__ bias, u16* __restrict__ out_bf,
               const float* __restrict__ resid, float* __restrict__ out_f32)
{
    __shared__ u16 At[128 * 32];
    __shared__ u16 Bt[128 * 32];   // stored [n][k]
    const int tid  = threadIdx.x;
    const int lane = tid & 31, wid = tid >> 5;
    const int hf   = lane >> 4, col = lane & 15;
    const int wm   = wid & 1,  wn  = wid >> 1;
    const int m0 = blockIdx.y * 128, n0 = blockIdx.x * 128;
    const int KROW = CINC * 9;
    const int CSTEPS = CINC / 32;

    v8f acc[4][2];
    for (int i = 0; i < 4; ++i)
        for (int j = 0; j < 2; ++j)
            acc[i][j] = (v8f){0.f,0.f,0.f,0.f,0.f,0.f,0.f,0.f};

    const unsigned atBase = lds_off(At);

    for (int ks = 0; ks < 9 * CSTEPS; ++ks) {
        const int tap = ks / CSTEPS;
        const int ci0 = (ks % CSTEPS) * 32;
        const int r = tap / 3 - 1, s = tap % 3 - 1;
        // A tile 128x32: async global->LDS copy (2 x 16B per thread)
        {
            int row = tid >> 1, coff = (tid & 1) * 16;
            const u16* src = wt + (size_t)(m0 + row) * KROW + tap * CINC + ci0 + coff;
            unsigned dst = atBase + (unsigned)(row * 32 + coff) * 2u;
            async_copy_b128(dst,      src);
            async_copy_b128(dst + 16, src + 8);
        }
        // B tile gather (im2col), stored [n][k]
        {
            int kk = tid & 31, nb = (tid >> 5) * 16;
            for (int v = 0; v < 16; ++v) {
                int ng = n0 + nb + v;
                int bb = ng / 9216, hw = ng % 9216, h = hw / 96, w = hw % 96;
                int hh = h + r, ww = w + s;
                u16 val = 0;
                if ((unsigned)hh < 96u && (unsigned)ww < 96u)
                    val = inp[(((size_t)bb * CINC + ci0 + kk) * 96 + hh) * 96 + ww];
                Bt[(nb + v) * 32 + kk] = val;
            }
        }
        async_wait0();
        __syncthreads();
        Frag a[4], bf[2];
        for (int i = 0; i < 4; ++i) a[i]  = load_a(At, 32, wm * 64 + i * 16 + col, 0, hf);
        for (int j = 0; j < 2; ++j) bf[j] = load_b(Bt, 32, wn * 32 + j * 16 + col, 0, hf);
        for (int i = 0; i < 4; ++i)
            for (int j = 0; j < 2; ++j)
                acc[i][j] = wmma_bf16(a[i], bf[j], acc[i][j]);
        __syncthreads();
    }
    // epilogue: C layout lane(N=col, half) holds M = base + hf*8 + rr
    for (int i = 0; i < 4; ++i)
        for (int j = 0; j < 2; ++j) {
            int Mg = m0 + wm * 64 + i * 16 + hf * 8;
            int Ng = n0 + wn * 32 + j * 16 + col;
            int bb = Ng / 9216, hw = Ng % 9216, h = hw / 96, w = hw % 96;
            for (int rr = 0; rr < 8; ++rr) {
                float v = acc[i][j][rr] + bias[Mg + rr];
                size_t oi = (((size_t)bb * MC + Mg + rr) * 96 + h) * 96 + w;
                if (MODE == 0) { v = fmaxf(v, 0.f); out_bf[oi] = f2bf(v); }
                else           { out_f32[oi] = resid[oi] + v; }
            }
        }
}

// ---------- windowed attention: one WG per (window, head) ----------
// y: bf16 [2][768][96][96]; out obf: bf16 [2][256][96][96]
__global__ __launch_bounds__(256)
void attn_kernel(const u16* __restrict__ y, u16* __restrict__ obf)
{
    extern __shared__ u16 smem[];
    u16* Kt = smem;                // [1024 i][64 c]   (A for S)  bytes [0, 131072)
    u16* Vs = smem + 65536;        // [64 c][1024 i]   (A for O)  bytes [131072, 262144)
    const int tid  = threadIdx.x;
    const int lane = tid & 31, wid = tid >> 5;
    const int hf   = lane >> 4, col = lane & 15;
    u16* qt   = smem + 131072 + wid * 1024;  // [16 j][64 c]  (B for S)
    u16* pscr = smem + 139264 + wid * 512;   // [16 j][32 i]  (B for O)

    const int head = blockIdx.x & 3;
    const int win  = blockIdx.x >> 2;
    const int bI   = win / 9, wy = (win % 9) / 3, wx = win % 3;
    const int h0 = wy * 32, w0 = wx * 32;
    const size_t ybase = (size_t)bI * 768 * 9216;
    const size_t ck = (size_t)head * 64;

    // ---- V panel via Tensor Data Mover: one descriptor, iterated over 64 channels.
    // Tile 32x32 halfs (rows stride 96), per-iter: global += 9216 elems, LDS += 1024 elems.
    if (wid == 0) {
        const u16* vg = y + ybase + (256 + ck) * 9216 + (size_t)h0 * 96 + w0;
        unsigned long long ga = (unsigned long long)vg;
        u32x4 g0; u32x8 g1; u32x4 g2; u32x4 g3;
        g0[0] = 1u;                                   // count=1
        g0[1] = 131072u;                              // lds_addr (bytes) = Vs
        g0[2] = (unsigned)ga;                         // global_addr lo
        g0[3] = (unsigned)(ga >> 32) | 0x80000000u;   // global_addr hi | type=2
        const unsigned td0 = 96u, td1 = 0x100000u;
        g1[0] = (1u << 16) | (1u << 19);              // data_size=2B, iterate_enable
        g1[1] = (td0 & 0xFFFFu) << 16;                // tensor_dim0 lo16
        g1[2] = (td0 >> 16) | ((td1 & 0xFFFFu) << 16);// dim0 hi16 | dim1 lo16
        g1[3] = (td1 >> 16) | (32u << 16);            // dim1 hi16 | tile_dim0=32
        g1[4] = 32u;                                  // tile_dim1=32, tile_dim2=0
        g1[5] = 96u;                                  // tensor_dim0_stride lo32 = 96
        g1[6] = 0u; g1[7] = 0u;
        g2[0] = 0u;                                   // tensor_dim2 (unused)
        g2[1] = 1024u;                                // lds_addr_increment (elems)
        g2[2] = 9216u;                                // global_addr_increment lo (elems)
        g2[3] = (63u << 16);                          // iterate_count = 63 (-> 64 iters)
        g3[0] = 0u; g3[1] = 0u; g3[2] = 0u; g3[3] = 0u;
        asm volatile("tensor_load_to_lds %0, %1, %2, %3"
                     :: "s"(g0), "s"(g1), "s"(g2), "s"(g3) : "memory");
        __builtin_amdgcn_s_wait_tensorcnt(0);
    }

    // cooperative K load with transpose (coalesced along window rows)
    for (int idx = tid; idx < 65536; idx += 256) {
        int c = idx >> 10, i = idx & 1023;
        int ph = i >> 5, pw = i & 31;
        Kt[i * 64 + c] = y[ybase + (ck + c) * 9216 + (size_t)(h0 + ph) * 96 + (w0 + pw)];
    }
    __syncthreads();

    for (int jt8 = 0; jt8 < 8; ++jt8) {
        const int jt = wid * 8 + jt8, j0 = jt * 16;
        // stage Q tile (per-wave): lane -> j=col, c in [hf*32, hf*32+32)
        {
            int jg = j0 + col;
            size_t sp = (size_t)(h0 + (jg >> 5)) * 96 + (w0 + (jg & 31));
            int cb = hf * 32;
            for (int cc = 0; cc < 32; ++cc)
                qt[col * 64 + cb + cc] = y[ybase + (512 + ck + cb + cc) * 9216 + sp];
        }
        lds_wave_fence();

        float m = -1e30f, l = 0.f;
        v8f oacc[4];
        for (int t = 0; t < 4; ++t) oacc[t] = (v8f){0.f,0.f,0.f,0.f,0.f,0.f,0.f,0.f};

        for (int ic = 0; ic < 32; ++ic) {
            const int i0 = ic * 32;
            // S = K^T Q for rows [i0,i0+32), cols j-tile; k-dim c=64 (2 steps)
            v8f s0 = (v8f){0.f,0.f,0.f,0.f,0.f,0.f,0.f,0.f};
            v8f s1 = s0;
            {
                Frag a0 = load_a(Kt, 64, i0 + col,      0, hf);
                Frag a1 = load_a(Kt, 64, i0 + 16 + col, 0, hf);
                Frag bq = load_b(qt, 64, col,           0, hf);
                s0 = wmma_bf16(a0, bq, s0);
                s1 = wmma_bf16(a1, bq, s1);
                a0 = load_a(Kt, 64, i0 + col,      32, hf);
                a1 = load_a(Kt, 64, i0 + 16 + col, 32, hf);
                bq = load_b(qt, 64, col,           32, hf);
                s0 = wmma_bf16(a0, bq, s0);
                s1 = wmma_bf16(a1, bq, s1);
            }
            // online softmax over i (columns live per-lane; rows split across half)
            float cm = -1e30f;
            for (int r = 0; r < 8; ++r) {
                s0[r] *= 0.125f; s1[r] *= 0.125f;
                cm = fmaxf(cm, fmaxf(s0[r], s1[r]));
            }
            cm = fmaxf(cm, __shfl_xor(cm, 16, 32));
            float mnew  = fmaxf(m, cm);
            float alpha = __expf(m - mnew);
            float psum = 0.f;
            for (int r = 0; r < 8; ++r) {
                float p0 = __expf(s0[r] - mnew);
                float p1 = __expf(s1[r] - mnew);
                psum += p0 + p1;
                pscr[col * 32 + hf * 8 + r]      = f2bf(p0);   // tile0 rows
                pscr[col * 32 + 16 + hf * 8 + r] = f2bf(p1);   // tile1 rows
            }
            psum += __shfl_xor(psum, 16, 32);
            l = l * alpha + psum;
            m = mnew;
            for (int t = 0; t < 4; ++t)
                for (int r = 0; r < 8; ++r) oacc[t][r] *= alpha;
            lds_wave_fence();
            // O += V[:, i0:i0+32] * P
            Frag bp = load_b(pscr, 32, col, 0, hf);
            for (int t = 0; t < 4; ++t) {
                Frag av = load_a(Vs, 1024, t * 16 + col, i0, hf);
                oacc[t] = wmma_bf16(av, bp, oacc[t]);
            }
            lds_wave_fence();   // before pscr is overwritten next chunk
        }
        // write O/l  -> obf[b][head*64 + c][h][w]
        float inv = 1.0f / l;
        int jg = j0 + col;
        size_t sp = (size_t)(h0 + (jg >> 5)) * 96 + (w0 + (jg & 31));
        for (int t = 0; t < 4; ++t)
            for (int r = 0; r < 8; ++r) {
                int cg = head * 64 + t * 16 + hf * 8 + r;
                obf[((size_t)bI * 256 + cg) * 9216 + sp] = f2bf(oacc[t][r] * inv);
            }
    }
}

// ---------- launch ----------
extern "C" void kernel_launch(void* const* d_in, const int* in_sizes, int n_in,
                              void* d_out, int out_size, void* d_ws, size_t ws_size,
                              hipStream_t stream) {
    const float* x  = (const float*)d_in[0];
    const float* W1 = (const float*)d_in[1];
    const float* b1 = (const float*)d_in[2];
    const float* W2 = (const float*)d_in[3];
    const float* b2 = (const float*)d_in[4];
    float* out = (float*)d_out;

    u16* xh  = (u16*)d_ws;                 // 2*512*96*96   = 9437184
    u16* w1h = xh  + 9437184;              // 768*4608      = 3538944
    u16* w2h = w1h + 3538944;              // 512*2304      = 1179648
    u16* ybf = w2h + 1179648;              // 2*768*96*96   = 14155776
    u16* obf = ybf + 14155776;             // 2*256*96*96   = 4718592  (total ~63 MB)

    k_cvt  <<<(9437184 + 255) / 256, 256, 0, stream>>>(x, xh, 9437184);
    k_cvt_w<<<(3538944 + 255) / 256, 256, 0, stream>>>(W1, w1h, 768, 512);
    k_cvt_w<<<(1179648 + 255) / 256, 256, 0, stream>>>(W2, w2h, 512, 256);

    conv_gemm<512, 768, 0><<<dim3(144, 6), 256, 0, stream>>>(xh, w1h, b1, ybf, nullptr, nullptr);

    attn_kernel<<<72, 256, 286720, stream>>>(ybf, obf);   // 280 KB LDS (K^T + V + Q/P scratch)

    conv_gemm<256, 512, 1><<<dim3(144, 4), 256, 0, stream>>>(obf, w2h, b2, nullptr, x, out);
}
